// AxialAttention_dynamic_17617955848893
// MI455X (gfx1250) — matrix operand order repaired and checked by reading, not compiled, not run on hardware
//
#include <hip/hip_runtime.h>
#include <hip/hip_bf16.h>
#include <stddef.h>
#include <stdint.h>

// ---- problem constants ----
#define GROUPS 8
#define Cc     512         // GROUPS*GP
#define Nn     16
#define Bb     1024        // N*W
#define TWO_C  1024
#define HW     4096        // H*W
#define EPSV   1e-5f
#define F_QR   0.1f
#define F_KR   0.1f
#define F_SVE  0.1f
#define F_SV   1.0f

typedef __bf16 v16bf __attribute__((ext_vector_type(16)));
typedef __bf16 v8bf  __attribute__((ext_vector_type(8)));
typedef float  v8f   __attribute__((ext_vector_type(8)));
typedef unsigned int u32x4 __attribute__((ext_vector_type(4)));
typedef int          i32x8 __attribute__((ext_vector_type(8)));
typedef int          i32x4 __attribute__((ext_vector_type(4)));

#ifndef __has_builtin
#define __has_builtin(x) 0
#endif
#if __has_builtin(__builtin_amdgcn_tensor_load_to_lds) && __has_builtin(__builtin_amdgcn_s_wait_tensorcnt)
#define USE_TDM 1
#else
#define USE_TDM 0
#endif

__device__ inline v8bf ld8(const __bf16* p) { return *(const v8bf*)p; }
__device__ inline v16bf cat8(v8bf lo, v8bf hi) {
    return __builtin_shufflevector(lo, hi, 0,1,2,3,4,5,6,7,8,9,10,11,12,13,14,15);
}

#if USE_TDM
// Tensor DMA Descriptor (D#) builder: 2D/3D bf16 tile, global -> LDS.
// dims/strides in elements (data_size = 2B). Issued per-wave, EXEC ignored.
// 6-arg builtin form (clang-23 / therock-10.0): trailing int32x8 zero-filled.
__device__ inline void tdm_load_bf16(uint32_t lds_addr, const void* gptr,
                                     uint32_t dim0, uint32_t dim1, uint32_t dim2,
                                     uint32_t t0, uint32_t t1, uint32_t t2,
                                     uint64_t stride0, uint64_t stride1) {
    uint64_t ga = (uint64_t)(uintptr_t)gptr;
    u32x4 g0;
    g0[0] = 1u;                                        // count=1, user mode
    g0[1] = lds_addr;                                  // LDS byte offset
    g0[2] = (uint32_t)ga;                              // global addr lo
    g0[3] = (uint32_t)((ga >> 32) & 0x1FFFFFFu) | (2u << 30);  // addr hi | type=2
    i32x8 g1;
    g1[0] = (int)(1u << 16);                           // wg_mask=0, data_size=1 (2B)
    g1[1] = (int)((dim0 & 0xFFFFu) << 16);             // tensor_dim0 lo
    g1[2] = (int)((dim0 >> 16) | ((dim1 & 0xFFFFu) << 16));
    g1[3] = (int)((dim1 >> 16) | (t0 << 16));          // tile_dim0
    g1[4] = (int)((t1 & 0xFFFFu) | (t2 << 16));        // tile_dim1 | tile_dim2
    g1[5] = (int)(uint32_t)stride0;                    // tensor_dim0_stride lo
    g1[6] = (int)(((uint32_t)(stride0 >> 32) & 0xFFFFu) |
                  ((uint32_t)(stride1 & 0xFFFFu) << 16));
    g1[7] = (int)(uint32_t)(stride1 >> 16);            // tensor_dim1_stride hi
    i32x4 g2; g2[0] = (int)dim2; g2[1] = 0; g2[2] = 0; g2[3] = 0;
    i32x4 g3; g3[0] = 0; g3[1] = 0; g3[2] = 0; g3[3] = 0;
    i32x8 g4; g4[0] = 0; g4[1] = 0; g4[2] = 0; g4[3] = 0;
              g4[4] = 0; g4[5] = 0; g4[6] = 0; g4[7] = 0;
    __builtin_amdgcn_tensor_load_to_lds(g0, g1, g2, g3, g4, 0);
}
#endif

// ---------------------------------------------------------------------------
// Kernel 0a: w_qkv fp32 -> bf16
// ---------------------------------------------------------------------------
__global__ __launch_bounds__(256) void prep_kernel(const float* __restrict__ w,
                                                   __bf16* __restrict__ wb) {
    int idx = blockIdx.x * 256 + threadIdx.x;
    if (idx < TWO_C * Cc) wb[idx] = (__bf16)w[idx];
}

// ---------------------------------------------------------------------------
// Kernel 0b: xT[n][hw][c] = bf16(x[n][c][hw])   (tiled LDS transpose)
// grid = (64, 8, 16), block = 256
// ---------------------------------------------------------------------------
__global__ __launch_bounds__(256) void xpose_kernel(const float* __restrict__ x,
                                                    __bf16* __restrict__ xT) {
    __shared__ __bf16 t[64][65];
    const int n = blockIdx.z, c0 = blockIdx.y * 64, hw0 = blockIdx.x * 64;
    const int tid = threadIdx.x;
    #pragma unroll
    for (int p = 0; p < 16; ++p) {
        int idx = p * 256 + tid;
        int c = idx >> 6, hw = idx & 63;
        t[c][hw] = (__bf16)x[((size_t)n * Cc + c0 + c) * HW + hw0 + hw];
    }
    __syncthreads();
    #pragma unroll
    for (int p = 0; p < 16; ++p) {
        int idx = p * 256 + tid;
        int hw = idx >> 6, c = idx & 63;
        xT[((size_t)n * HW + hw0 + hw) * Cc + c0 + c] = t[c][hw];
    }
}

// ---------------------------------------------------------------------------
// Kernel 1: qkv[n](1024 x 4096) = W(1024x512) x x[n](512x4096), BN, bf16 out
// grid = (64 hw-tiles, 16 o-tiles, 16 n), block = 256.  TDM double-buffered.
// ---------------------------------------------------------------------------
__global__ __launch_bounds__(256) void qkv_gemm_kernel(
    const __bf16* __restrict__ xT,
    const __bf16* __restrict__ wb,
    const float*  __restrict__ gma, const float* __restrict__ bta,
    const float*  __restrict__ mu,  const float* __restrict__ var,
    __bf16* __restrict__ qkv) {

    __shared__ __align__(16) __bf16 sA[2][64 * 32];   // [m][k]
    __shared__ __align__(16) __bf16 sBt[2][64 * 32];  // [nn][k]

    const int n   = blockIdx.z;
    const int o0  = blockIdx.y * 64;
    const int hw0 = blockIdx.x * 64;
    const int tid = threadIdx.x;
    const int lane = tid & 31, wv = tid >> 5;
    const int hf = lane >> 4, l15 = lane & 15;
    const int t0 = wv * 2, t1 = t0 + 1;
    const int ti0 = t0 >> 2, tj0 = t0 & 3, ti1 = t1 >> 2, tj1 = t1 & 3;

    const __bf16* gA = wb + (size_t)o0 * Cc;                      // 64 rows x 512
    const __bf16* gB = xT + ((size_t)n * HW + hw0) * Cc;          // 64 rows x 512

#if !USE_TDM
    const int cm = tid >> 2, ck = (tid & 3) * 8;                  // 16B chunk per thread
#endif

    auto stage = [&](int buf, int k0) {
#if USE_TDM
        if (wv == 0) {
            tdm_load_bf16((uint32_t)(uintptr_t)&sA[buf][0],  gA + k0,
                          Cc - k0, 64, 0, 32, 64, 0, Cc, 0);
            tdm_load_bf16((uint32_t)(uintptr_t)&sBt[buf][0], gB + k0,
                          Cc - k0, 64, 0, 32, 64, 0, Cc, 0);
        }
#else
        *(v8bf*)&sA[buf][cm * 32 + ck]  = *(const v8bf*)(gA + (size_t)cm * Cc + k0 + ck);
        *(v8bf*)&sBt[buf][cm * 32 + ck] = *(const v8bf*)(gB + (size_t)cm * Cc + k0 + ck);
#endif
    };

    v8f acc0 = {}, acc1 = {};
    stage(0, 0);
    for (int k0 = 0; k0 < Cc; k0 += 32) {
        int buf = (k0 >> 5) & 1;
#if USE_TDM
        __builtin_amdgcn_s_wait_tensorcnt(0);
#endif
        __syncthreads();
        if (k0 + 32 < Cc) stage(buf ^ 1, k0 + 32);   // overlaps with WMMA below

        const __bf16* pa0 = &sA[buf][(ti0 * 16 + l15) * 32 + hf * 8];
        const __bf16* pa1 = &sA[buf][(ti1 * 16 + l15) * 32 + hf * 8];
        const __bf16* pb0 = &sBt[buf][(tj0 * 16 + l15) * 32 + hf * 8];
        const __bf16* pb1 = &sBt[buf][(tj1 * 16 + l15) * 32 + hf * 8];
        v16bf a0 = cat8(ld8(pa0), ld8(pa0 + 16));
        v16bf a1 = cat8(ld8(pa1), ld8(pa1 + 16));
        v16bf b0 = cat8(ld8(pb0), ld8(pb0 + 16));
        v16bf b1 = cat8(ld8(pb1), ld8(pb1 + 16));
        acc0 = __builtin_amdgcn_wmma_f32_16x16x32_bf16(false, a0, false, b0,
                                                       (short)0, acc0, false, false);
        acc1 = __builtin_amdgcn_wmma_f32_16x16x32_bf16(false, a1, false, b1,
                                                       (short)0, acc1, false, false);
    }

    // BN epilogue + bf16 store (D layout: n = l15, m = r + 8*hf)
    #pragma unroll
    for (int r = 0; r < 8; ++r) {
        int o  = o0 + ti0 * 16 + r + hf * 8;
        int hw = hw0 + tj0 * 16 + l15;
        float a = gma[o] * rsqrtf(var[o] + EPSV);
        qkv[((size_t)n * TWO_C + o) * HW + hw] = (__bf16)(a * (acc0[r] - mu[o]) + bta[o]);
    }
    #pragma unroll
    for (int r = 0; r < 8; ++r) {
        int o  = o0 + ti1 * 16 + r + hf * 8;
        int hw = hw0 + tj1 * 16 + l15;
        float a = gma[o] * rsqrtf(var[o] + EPSV);
        qkv[((size_t)n * TWO_C + o) * HW + hw] = (__bf16)(a * (acc1[r] - mu[o]) + bta[o]);
    }
}

// ---------------------------------------------------------------------------
// Kernel 2: attention per (b, g).  grid = (1024, 8), block = 256 (8 waves)
// ---------------------------------------------------------------------------
__global__ __launch_bounds__(256) void attn_kernel(
    const __bf16* __restrict__ qkv,
    const float*  __restrict__ rel,       // (128 x 127)
    const float* __restrict__ sg, const float* __restrict__ sb,
    const float* __restrict__ sm, const float* __restrict__ sv_,
    const float* __restrict__ og, const float* __restrict__ ob,
    const float* __restrict__ om, const float* __restrict__ ov,
    float* __restrict__ out) {

    __shared__ __align__(16) __bf16 lq[64 * 32];      // [i][c]
    __shared__ __align__(16) __bf16 lk[64 * 32];      // [j][c]
    __shared__ __align__(16) __bf16 lv[64 * 64];      // [c][j]
    __shared__ __align__(16) __bf16 lrq[127 * 32];    // q_emb^T  [d][c]
    __shared__ __align__(16) __bf16 lrk[127 * 32];    // k_emb^T  [d][c]
    __shared__ __align__(16) float  lsim[64 * 64];    // sim[i][j] fp32
    __shared__ __align__(16) __bf16 lsimb[64 * 64];   // sim[i][j] bf16

    const int b = blockIdx.x, g = blockIdx.y;
    const int n = b >> 6, w = b & 63;
    const int tid = threadIdx.x;
    const int lane = tid & 31, wv = tid >> 5;
    const int hf = lane >> 4, l15 = lane & 15;

    const __bf16* base = qkv + ((size_t)(n * TWO_C + g * 128)) * HW + w;

#if USE_TDM
    // v tile (64 ch x 64 h, single-element lines along fixed w) as a 3D TDM tile
    if (wv == 0)
        tdm_load_bf16((uint32_t)(uintptr_t)&lv[0], base + (size_t)64 * HW,
                      1, 64, 64, 1, 64, 64, 64, HW);
#else
    for (int idx = tid; idx < 64 * 64; idx += 256) {
        int cc = idx >> 6, h = idx & 63;
        lv[cc * 64 + h] = base[(size_t)(64 + cc) * HW + h * 64];
    }
#endif
    // q,k staged transposed ([h][c]) so WMMA fragments + qr/kr dots are contiguous
    for (int idx = tid; idx < 64 * 64; idx += 256) {
        int cc = idx >> 6, h = idx & 63;
        __bf16 val = base[(size_t)cc * HW + h * 64];
        if (cc < 32) lq[h * 32 + cc] = val;
        else         lk[h * 32 + (cc - 32)] = val;
    }
    // relative rows 0..63, transposed to [d][c]
    for (int idx = tid; idx < 127 * 32; idx += 256) {
        int d = idx >> 5, c = idx & 31;
        lrq[idx] = (__bf16)rel[c * 127 + d];
        lrk[idx] = (__bf16)rel[(32 + c) * 127 + d];
    }
#if USE_TDM
    __builtin_amdgcn_s_wait_tensorcnt(0);
#endif
    __syncthreads();

    // ---- qk = q^T k (M=i, N=j, K=c=32): 16 tiles, 2 per wave ----
    #pragma unroll
    for (int tt = 0; tt < 2; ++tt) {
        int t = wv * 2 + tt, ti = t >> 2, tj = t & 3;
        const __bf16* pa = &lq[(ti * 16 + l15) * 32 + hf * 8];
        const __bf16* pb = &lk[(tj * 16 + l15) * 32 + hf * 8];
        v16bf a  = cat8(ld8(pa), ld8(pa + 16));
        v16bf bm = cat8(ld8(pb), ld8(pb + 16));
        v8f d = {};
        d = __builtin_amdgcn_wmma_f32_16x16x32_bf16(false, a, false, bm,
                                                    (short)0, d, false, false);
        #pragma unroll
        for (int r = 0; r < 8; ++r)
            lsim[(ti * 16 + r + hf * 8) * 64 + tj * 16 + l15] = d[r];
    }
    __syncthreads();

    // ---- qr/kr (contiguous bf16 dots) + BN-combine of the 3 stacked channels --
    float aqk = sg[g]      * rsqrtf(sv_[g]      + EPSV);
    float bqk = sb[g]      - sm[g]      * aqk;
    float aqr = sg[8 + g]  * rsqrtf(sv_[8 + g]  + EPSV);
    float bqr = sb[8 + g]  - sm[8 + g]  * aqr;
    float akr = sg[16 + g] * rsqrtf(sv_[16 + g] + EPSV);
    float bkr = sb[16 + g] - sm[16 + g] * akr;

    #pragma unroll 1
    for (int q = 0; q < 16; ++q) {
        int idx = tid * 16 + q;
        int i = idx >> 6, j = idx & 63;
        int d1 = i - j + 63, d2 = j - i + 63;
        const v8bf* q8 = (const v8bf*)&lq[i * 32];
        const v8bf* r8 = (const v8bf*)&lrq[d1 * 32];
        const v8bf* k8 = (const v8bf*)&lk[j * 32];
        const v8bf* s8 = (const v8bf*)&lrk[d2 * 32];
        float qr = 0.f, kr = 0.f;
        #pragma unroll
        for (int c8 = 0; c8 < 4; ++c8) {
            v8bf qa = q8[c8], ra = r8[c8], ka = k8[c8], sa = s8[c8];
            #pragma unroll
            for (int e = 0; e < 8; ++e) {
                qr += (float)qa[e] * (float)ra[e];
                kr += (float)ka[e] * (float)sa[e];
            }
        }
        lsim[idx] = aqk * lsim[idx] + bqk
                  + aqr * (F_QR * qr) + bqr
                  + akr * (F_KR * kr) + bkr;
    }
    __syncthreads();

    // ---- softmax over j: 4 lanes per row, shfl_xor reductions ----
    {
        int row = tid >> 2, seg = tid & 3;
        float* rp = &lsim[row * 64 + seg * 16];
        float mx = rp[0];
        #pragma unroll
        for (int j = 1; j < 16; ++j) mx = fmaxf(mx, rp[j]);
        mx = fmaxf(mx, __shfl_xor(mx, 1, 4));
        mx = fmaxf(mx, __shfl_xor(mx, 2, 4));
        float s = 0.f;
        #pragma unroll
        for (int j = 0; j < 16; ++j) { float e = __expf(rp[j] - mx); rp[j] = e; s += e; }
        s += __shfl_xor(s, 1, 4);
        s += __shfl_xor(s, 2, 4);
        float inv = 1.f / s;
        __bf16* bp = &lsimb[row * 64 + seg * 16];
        #pragma unroll
        for (int j = 0; j < 16; ++j) { float e = rp[j] * inv; rp[j] = e; bp[j] = (__bf16)e; }
    }
    __syncthreads();

    // ---- sv = v . sim^T (M=c, N=i, K=j=64) + sve + out-BN + store ----
    #pragma unroll
    for (int tt = 0; tt < 2; ++tt) {
        int t = wv * 2 + tt, tc = t >> 2, tii = t & 3;
        v8f d = {};
        #pragma unroll
        for (int kc = 0; kc < 2; ++kc) {
            const __bf16* pa = &lv[(tc * 16 + l15) * 64 + kc * 32 + hf * 8];
            const __bf16* pb = &lsimb[(tii * 16 + l15) * 64 + kc * 32 + hf * 8];
            v16bf a  = cat8(ld8(pa), ld8(pa + 16));
            v16bf bm = cat8(ld8(pb), ld8(pb + 16));
            d = __builtin_amdgcn_wmma_f32_16x16x32_bf16(false, a, false, bm,
                                                        (short)0, d, false, false);
        }
        #pragma unroll 1
        for (int r = 0; r < 8; ++r) {
            int c = tc * 16 + r + hf * 8;        // channel within group (0..63)
            int i = tii * 16 + l15;
            const float* vrow = rel + (64 + c) * 127 + (i + 63);
            float sve = 0.f;
            #pragma unroll 8
            for (int j = 0; j < 64; ++j) sve += lsim[i * 64 + j] * vrow[-j];
            int o0 = g * 128 + 2 * c;
            float a0 = og[o0]     * rsqrtf(ov[o0]     + EPSV);
            float c0 = ob[o0]     - om[o0]     * a0;
            float a1 = og[o0 + 1] * rsqrtf(ov[o0 + 1] + EPSV);
            float c1 = ob[o0 + 1] - om[o0 + 1] * a1;
            float res = a0 * (F_SV * d[r]) + c0 + a1 * (F_SVE * sve) + c1;
            out[(((size_t)n * Cc + g * 64 + c) * 64 + i) * 64 + w] = res;
        }
    }
}

// ---------------------------------------------------------------------------
extern "C" void kernel_launch(void* const* d_in, const int* in_sizes, int n_in,
                              void* d_out, int out_size, void* d_ws, size_t ws_size,
                              hipStream_t stream) {
    const float* x        = (const float*)d_in[0];
    const float* w_qkv    = (const float*)d_in[1];
    const float* relative = (const float*)d_in[2];
    const float* qkv_g = (const float*)d_in[3];
    const float* qkv_b = (const float*)d_in[4];
    const float* qkv_m = (const float*)d_in[5];
    const float* qkv_v = (const float*)d_in[6];
    const float* sim_g = (const float*)d_in[7];
    const float* sim_b = (const float*)d_in[8];
    const float* sim_m = (const float*)d_in[9];
    const float* sim_v = (const float*)d_in[10];
    const float* out_g = (const float*)d_in[11];
    const float* out_b = (const float*)d_in[12];
    const float* out_m = (const float*)d_in[13];
    const float* out_v = (const float*)d_in[14];
    float* out = (float*)d_out;

    // workspace: [0,1MB) w bf16 ; [1MB,65MB) xT bf16 ; [65MB,193MB) qkv bf16
    __bf16* wb  = (__bf16*)d_ws;
    __bf16* xT  = (__bf16*)((char*)d_ws + ((size_t)1 << 20));
    __bf16* qkv = (__bf16*)((char*)d_ws + ((size_t)65 << 20));

    prep_kernel<<<(TWO_C * Cc + 255) / 256, 256, 0, stream>>>(w_qkv, wb);
    xpose_kernel<<<dim3(HW / 64, Cc / 64, Nn), 256, 0, stream>>>(x, xT);
    qkv_gemm_kernel<<<dim3(HW / 64, TWO_C / 64, Nn), 256, 0, stream>>>(
        xT, wb, qkv_g, qkv_b, qkv_m, qkv_v, qkv);
    attn_kernel<<<dim3(Bb, GROUPS), 256, 0, stream>>>(
        qkv, relative, sim_g, sim_b, sim_m, sim_v,
        out_g, out_b, out_m, out_v, out);
}
